// ANIMA1_86887188398437
// MI455X (gfx1250) — compile-verified, matrix operand order checked by reading, not run on hardware
//
#include <hip/hip_runtime.h>
#include <hip/hip_bf16.h>
#include <math.h>
#include <stdint.h>

typedef _Float16 h16;
typedef __attribute__((ext_vector_type(16))) _Float16 v16h;
typedef __attribute__((ext_vector_type(8)))  float    v8f;

#define DEVF __device__ __forceinline__

#define TSEQ 2048
#define BATCH 64
// padded LDS row strides (elements) to kill bank conflicts on fragment loads
#define RP256 264
#define RP768 776
#define RP128 136
#define RPF 260

union F16x16 { v16h v; h16 h[16]; };

// A/B fragment: 8 contiguous halves at p[0..7] and 8 at p[16..23]
DEVF v16h ldfrag(const h16* p) {
  F16x16 u;
#pragma unroll
  for (int i = 0; i < 8; ++i) { u.h[i] = p[i]; u.h[8 + i] = p[16 + i]; }
  return u.v;
}
DEVF v16h ldfragf(const float* p) {
  F16x16 u;
#pragma unroll
  for (int i = 0; i < 8; ++i) { u.h[i] = (h16)p[i]; u.h[8 + i] = (h16)p[16 + i]; }
  return u.v;
}
DEVF v8f wmma16(v16h a, v16h b, v8f c) {
  return __builtin_amdgcn_wmma_f32_16x16x32_f16(false, a, false, b, (short)0, c, false, false);
}
// branchless fast activations: v_exp_f32 + v_rcp_f32, no exec divergence
DEVF float sigf(float x) {
  return __builtin_amdgcn_rcpf(1.0f + __expf(-x));
}
DEVF float tanh_fast(float x) {
  x = fminf(fmaxf(x, -15.0f), 15.0f);
  float e = __expf(2.0f * x);
  return (e - 1.0f) * __builtin_amdgcn_rcpf(e + 1.0f);
}
DEVF int waveid() { return __builtin_amdgcn_readfirstlane(threadIdx.x >> 5); }

// gfx1250 async DMA: global -> LDS, 16B per lane, tracked by ASYNCcnt
DEVF void async_g2l_b128(const h16* ldsDst, const h16* gSrc) {
  unsigned lo = (unsigned)(uintptr_t)ldsDst;   // low 32 bits of generic addr = LDS byte offset
  asm volatile("global_load_async_to_lds_b128 %0, %1, off"
               :: "v"(lo), "v"(gSrc) : "memory");
}
DEVF void wait_async0() { asm volatile("s_wait_asynccnt 0" ::: "memory"); }

// stage a [64][256] f16 activation slab into padded LDS rows (RP256)
DEVF void stage_slab(h16* dst, const h16* src) {
  for (int q = threadIdx.x; q < 2048; q += (int)blockDim.x) {
    int row = q >> 5, cq = (q & 31) << 3;
    async_g2l_b128(dst + row * RP256 + cq, src + row * 256 + cq);
  }
  wait_async0();
}

// ---------------- weight convert + transpose: dst[n*K+k] = (h16)src[k*N+n]
__global__ void k_wcvt(const float* __restrict__ src, h16* __restrict__ dst, int K, int N) {
  int i = blockIdx.x * blockDim.x + threadIdx.x;
  if (i < K * N) { int n = i / K; int k = i - n * K; dst[i] = (h16)src[(size_t)k * N + n]; }
}

// ---------------- encode: sensed = tanh(obs@We+b) -> X[T,B,256]; SIN = pad(tanh(sensed@Wc+b))
__global__ void __launch_bounds__(128) k_encode(
    const float* __restrict__ obs, const h16* __restrict__ WeT, const float* __restrict__ Web,
    const h16* __restrict__ WcT, const float* __restrict__ Wcb,
    h16* __restrict__ X, h16* __restrict__ SIN)
{
  __shared__ h16 sens[16 * RP256];
  int mt = blockIdx.x;            // 8192 tiles over T*B rows
  int t = mt >> 2;
  int b0 = (mt & 3) << 4;
  int lane = threadIdx.x & 31, wid = waveid();
  int lr = lane & 15, lko = (lane >> 4) << 3, cc = lane & 15, rb = (lane >> 4) << 3;

  const float* arow = obs + ((size_t)(b0 + lr) * TSEQ + t) * 64 + lko;
  for (int nt = wid; nt < 16; nt += 4) {
    int n0 = nt << 4;
    v8f acc = {};
#pragma unroll
    for (int kk = 0; kk < 64; kk += 32)
      acc = wmma16(ldfragf(arow + kk), ldfrag(WeT + (size_t)(n0 + lr) * 64 + kk + lko), acc);
#pragma unroll
    for (int i = 0; i < 8; ++i) {
      int row = rb + i, col = n0 + cc;
      float v = tanh_fast(acc[i] + Web[col]);
      sens[row * RP256 + col] = (h16)v;
      X[((size_t)t * 64 + b0 + row) * 256 + col] = (h16)v;
    }
  }
  __syncthreads();
  for (int nt = wid; nt < 8; nt += 4) {
    int n0 = nt << 4;
    v8f acc = {};
#pragma unroll
    for (int kk = 0; kk < 256; kk += 32)
      acc = wmma16(ldfrag(sens + lr * RP256 + kk + lko), ldfrag(WcT + (size_t)(n0 + lr) * 256 + kk + lko), acc);
#pragma unroll
    for (int i = 0; i < 8; ++i) {
      int row = rb + i, col = n0 + cc;
      SIN[((size_t)t * 64 + b0 + row) * 256 + col] = (h16)tanh_fast(acc[i] + Wcb[col]);
    }
  }
  for (int idx = threadIdx.x; idx < 16 * 128; idx += 128) {
    int row = idx >> 7, col = 128 + (idx & 127);
    SIN[((size_t)t * 64 + b0 + row) * 256 + col] = (h16)0.0f;
  }
}

// ---------------- GRU scan: persistent single workgroup, h lives in LDS
// LDS: hF[64][260]f32 | hH | rhH | zH | hnH | XtH  (f16 [64][264] each)
#define GRU_HF 0
#define GRU_HH 66560
#define GRU_RH 100352
#define GRU_ZH 134144
#define GRU_HN 167936
#define GRU_XT 201728
#define GRU_SMEM 235520
__global__ void __launch_bounds__(256) k_gru(
    const h16* __restrict__ X,
    const h16* __restrict__ WzT, const float* __restrict__ Wzb,
    const h16* __restrict__ WrT, const float* __restrict__ Wrb,
    const h16* __restrict__ WhT, const float* __restrict__ Whb,
    const h16* __restrict__ WpT, const float* __restrict__ Wpb,
    h16* __restrict__ M)
{
  extern __shared__ char smem[];
  float* hF = (float*)(smem + GRU_HF);
  h16* hH = (h16*)(smem + GRU_HH);
  h16* rhH = (h16*)(smem + GRU_RH);
  h16* zH = (h16*)(smem + GRU_ZH);
  h16* hnH = (h16*)(smem + GRU_HN);
  h16* XtH = (h16*)(smem + GRU_XT);

  int lane = threadIdx.x & 31, wid = waveid();
  int lr = lane & 15, lko = (lane >> 4) << 3, cc = lane & 15, rb = (lane >> 4) << 3;

  for (int i = threadIdx.x; i < 64 * RPF; i += 256) hF[i] = 0.0f;
  for (int i = threadIdx.x; i < 64 * RP256; i += 256) hH[i] = (h16)0.0f;
  __syncthreads();

  for (int t = 0; t < TSEQ; ++t) {
    const h16* Xt = X + (size_t)t * 64 * 256;
    bool prop = (((t + 1) & 15) == 0) && (t < TSEQ - 1);
    if (t + 1 < TSEQ) __builtin_prefetch(Xt + 64 * 256 + (threadIdx.x << 6), 0, 1);

    // async-DMA stage this step's input slab into LDS
    stage_slab(XtH, Xt);
    __syncthreads();

    // phase A: z and r (128 tiles), rh = r*h
    for (int tl = wid; tl < 128; tl += 8) {
      int gate = tl >> 6, tt = tl & 63;
      int m0 = (tt & 3) << 4, n0 = (tt >> 2) << 4;
      const h16* WT = gate ? WrT : WzT;
      v8f acc = {};
      for (int kk = 0; kk < 512; kk += 32) {
        const h16* ap = (kk < 256) ? (XtH + (m0 + lr) * RP256 + kk + lko)
                                   : (hH + (m0 + lr) * RP256 + (kk - 256) + lko);
        acc = wmma16(ldfrag(ap), ldfrag(WT + (size_t)(n0 + lr) * 512 + kk + lko), acc);
      }
#pragma unroll
      for (int i = 0; i < 8; ++i) {
        int row = m0 + rb + i, col = n0 + cc;
        float s = sigf(acc[i] + (gate ? Wrb[col] : Wzb[col]));
        if (gate) rhH[row * RP256 + col] = (h16)(s * hF[row * RPF + col]);
        else      zH[row * RP256 + col] = (h16)s;
      }
    }
    __syncthreads();
    // phase B: hc, h_new (64 tiles)
    for (int tl = wid; tl < 64; tl += 8) {
      int m0 = (tl & 3) << 4, n0 = (tl >> 2) << 4;
      v8f acc = {};
      for (int kk = 0; kk < 512; kk += 32) {
        const h16* ap = (kk < 256) ? (XtH + (m0 + lr) * RP256 + kk + lko)
                                   : (rhH + (m0 + lr) * RP256 + (kk - 256) + lko);
        acc = wmma16(ldfrag(ap), ldfrag(WhT + (size_t)(n0 + lr) * 512 + kk + lko), acc);
      }
#pragma unroll
      for (int i = 0; i < 8; ++i) {
        int row = m0 + rb + i, col = n0 + cc;
        float hc = tanh_fast(acc[i] + Whb[col]);
        float z = (float)zH[row * RP256 + col];
        float h = hF[row * RPF + col];
        float hn = (1.0f - z) * h + z * hc;
        M[((size_t)t * 64 + row) * 256 + col] = (h16)hn;
        hnH[row * RP256 + col] = (h16)hn;
        if (!prop) { hF[row * RPF + col] = hn; hH[row * RP256 + col] = (h16)hn; }
      }
    }
    __syncthreads();
    // phase C: chunk-boundary state propagation h = tanh(h_new@Wp+b)
    if (prop) {
      for (int tl = wid; tl < 64; tl += 8) {
        int m0 = (tl & 3) << 4, n0 = (tl >> 2) << 4;
        v8f acc = {};
        for (int kk = 0; kk < 256; kk += 32)
          acc = wmma16(ldfrag(hnH + (m0 + lr) * RP256 + kk + lko),
                       ldfrag(WpT + (size_t)(n0 + lr) * 256 + kk + lko), acc);
#pragma unroll
        for (int i = 0; i < 8; ++i) {
          int row = m0 + rb + i, col = n0 + cc;
          float v = tanh_fast(acc[i] + Wpb[col]);
          hF[row * RPF + col] = v;
          hH[row * RP256 + col] = (h16)v;
        }
      }
      __syncthreads();
    }
  }
}

// ---------------- interaction scan: persistent single workgroup, D carry in LDS
// LDS: gatedH [64][776] | DH [64][264] | t1H 3x[64][136] | StH [64][264] | MtH [64][264]
#define INT_GT 0
#define INT_DH 99328
#define INT_T1 133120
#define INT_ST 185344
#define INT_MT 219136
#define INT_SMEM 252928
#define T1SZ (64 * RP128)
__global__ void __launch_bounds__(256) k_inter(
    const h16* __restrict__ SIN, const h16* __restrict__ M,
    const h16* __restrict__ phiT, const float* __restrict__ phib,
    const h16* __restrict__ ScT, const float* __restrict__ Scb,
    const h16* __restrict__ SeT, const float* __restrict__ Seb,
    const h16* __restrict__ McT, const float* __restrict__ Mcb,
    const h16* __restrict__ MeT, const float* __restrict__ Meb,
    const h16* __restrict__ DcT, const float* __restrict__ Dcb,
    const h16* __restrict__ DeT, const float* __restrict__ Deb,
    h16* __restrict__ SSEQ, h16* __restrict__ M2SEQ, h16* __restrict__ DSEQ,
    float* __restrict__ outS, float* __restrict__ outM, float* __restrict__ outD)
{
  extern __shared__ char smem[];
  h16* gatedH = (h16*)(smem + INT_GT);
  h16* DH = (h16*)(smem + INT_DH);
  h16* t1H = (h16*)(smem + INT_T1);
  h16* StH = (h16*)(smem + INT_ST);
  h16* MtH = (h16*)(smem + INT_MT);

  int lane = threadIdx.x & 31, wid = waveid();
  int lr = lane & 15, lko = (lane >> 4) << 3, cc = lane & 15, rb = (lane >> 4) << 3;

  for (int i = threadIdx.x; i < 64 * RP256; i += 256) DH[i] = (h16)0.0f;
  __syncthreads();

  for (int t = 0; t < TSEQ; ++t) {
    const h16* St = SIN + (size_t)t * 64 * 256;
    const h16* Mt = M + (size_t)t * 64 * 256;
    if (t + 1 < TSEQ) __builtin_prefetch(St + 64 * 256 + (threadIdx.x << 6), 0, 1);

    // async-DMA stage S_in[t] and M[t] slabs into LDS
    stage_slab(StH, St);
    stage_slab(MtH, Mt);
    __syncthreads();

    // phase A: gated = comb * sigmoid(comb@phi+b), comb = [S|M|D]  (192 tiles)
    for (int tl = wid; tl < 192; tl += 8) {
      int m0 = (tl & 3) << 4, n0 = (tl >> 2) << 4;
      v8f acc = {};
      for (int kk = 0; kk < 768; kk += 32) {
        const h16* ap = (kk < 256) ? (StH + (m0 + lr) * RP256 + kk + lko)
                      : (kk < 512) ? (MtH + (m0 + lr) * RP256 + (kk - 256) + lko)
                                   : (DH + (m0 + lr) * RP256 + (kk - 512) + lko);
        acc = wmma16(ldfrag(ap), ldfrag(phiT + (size_t)(n0 + lr) * 768 + kk + lko), acc);
      }
#pragma unroll
      for (int i = 0; i < 8; ++i) {
        int row = m0 + rb + i, col = n0 + cc;
        float cv = (col < 256) ? (float)StH[row * RP256 + col]
                 : (col < 512) ? (float)MtH[row * RP256 + col - 256]
                               : (float)DH[row * RP256 + col - 512];
        gatedH[row * RP768 + col] = (h16)(cv * sigf(acc[i] + phib[col]));
      }
    }
    __syncthreads();
    // phase B: 3 compress heads t1 = tanh(gated@Xc+b)  (96 tiles)
    for (int tl = wid; tl < 96; tl += 8) {
      int head = tl / 32, tt = tl % 32;
      int m0 = (tt & 3) << 4, n0 = (tt >> 2) << 4;
      const h16* WT = head == 0 ? ScT : head == 1 ? McT : DcT;
      const float* bb = head == 0 ? Scb : head == 1 ? Mcb : Dcb;
      v8f acc = {};
      for (int kk = 0; kk < 768; kk += 32)
        acc = wmma16(ldfrag(gatedH + (m0 + lr) * RP768 + kk + lko),
                     ldfrag(WT + (size_t)(n0 + lr) * 768 + kk + lko), acc);
#pragma unroll
      for (int i = 0; i < 8; ++i)
        t1H[head * T1SZ + (m0 + rb + i) * RP128 + n0 + cc] = (h16)tanh_fast(acc[i] + bb[n0 + cc]);
    }
    __syncthreads();
    // phase C: 3 expand heads out = tanh(t1@Xe+b)  (192 tiles)
    for (int tl = wid; tl < 192; tl += 8) {
      int head = tl / 64, tt = tl % 64;
      int m0 = (tt & 3) << 4, n0 = (tt >> 2) << 4;
      const h16* WT = head == 0 ? SeT : head == 1 ? MeT : DeT;
      const float* bb = head == 0 ? Seb : head == 1 ? Meb : Deb;
      h16* dst = head == 0 ? SSEQ : head == 1 ? M2SEQ : DSEQ;
      float* fo = head == 0 ? outS : head == 1 ? outM : outD;
      v8f acc = {};
      for (int kk = 0; kk < 128; kk += 32)
        acc = wmma16(ldfrag(t1H + head * T1SZ + (m0 + lr) * RP128 + kk + lko),
                     ldfrag(WT + (size_t)(n0 + lr) * 128 + kk + lko), acc);
#pragma unroll
      for (int i = 0; i < 8; ++i) {
        int row = m0 + rb + i, col = n0 + cc;
        float v = tanh_fast(acc[i] + bb[col]);
        dst[((size_t)t * 64 + row) * 256 + col] = (h16)v;
        if (head == 2) DH[row * RP256 + col] = (h16)v;
        if (t == TSEQ - 1) fo[row * 256 + col] = v;
      }
    }
    __syncthreads();
  }
}

// ---------------- post: decision/actions + preds (fully parallel over T*B)
__global__ void __launch_bounds__(128) k_post(
    const h16* __restrict__ SSEQ, const h16* __restrict__ M2SEQ, const h16* __restrict__ DSEQ,
    const h16* __restrict__ dcT, const float* __restrict__ dcb,
    const h16* __restrict__ deT, const float* __restrict__ deb,
    const h16* __restrict__ ohT, const float* __restrict__ ohb,
    const h16* __restrict__ p1T, const float* __restrict__ p1b,
    const h16* __restrict__ p2T, const float* __restrict__ p2b,
    float* __restrict__ actions, float* __restrict__ preds)
{
  __shared__ h16 uH[16 * RP128];
  __shared__ h16 vH[16 * RP128];
  __shared__ h16 decH[16 * RP256];
  int mt = blockIdx.x;
  int t = mt >> 2, b0 = (mt & 3) << 4;
  int lane = threadIdx.x & 31, wid = waveid();
  int lr = lane & 15, lko = (lane >> 4) << 3, cc = lane & 15, rb = (lane >> 4) << 3;

  const h16* St = SSEQ + ((size_t)t * 64 + b0) * 256;
  const h16* Mt = M2SEQ + ((size_t)t * 64 + b0) * 256;
  const h16* Dt = DSEQ + ((size_t)t * 64 + b0) * 256;

  // u = tanh(combined@dc+b), K=768
  for (int nt = wid; nt < 8; nt += 4) {
    int n0 = nt << 4;
    v8f acc = {};
    for (int kk = 0; kk < 768; kk += 32) {
      const h16* ap = (kk < 256) ? (St + lr * 256 + kk + lko)
                    : (kk < 512) ? (Mt + lr * 256 + (kk - 256) + lko)
                                 : (Dt + lr * 256 + (kk - 512) + lko);
      acc = wmma16(ldfrag(ap), ldfrag(dcT + (size_t)(n0 + lr) * 768 + kk + lko), acc);
    }
#pragma unroll
    for (int i = 0; i < 8; ++i) uH[(rb + i) * RP128 + n0 + cc] = (h16)tanh_fast(acc[i] + dcb[n0 + cc]);
  }
  // v = tanh([S|M2]@p1+b), K=512
  for (int nt = wid; nt < 8; nt += 4) {
    int n0 = nt << 4;
    v8f acc = {};
    for (int kk = 0; kk < 512; kk += 32) {
      const h16* ap = (kk < 256) ? (St + lr * 256 + kk + lko)
                                 : (Mt + lr * 256 + (kk - 256) + lko);
      acc = wmma16(ldfrag(ap), ldfrag(p1T + (size_t)(n0 + lr) * 512 + kk + lko), acc);
    }
#pragma unroll
    for (int i = 0; i < 8; ++i) vH[(rb + i) * RP128 + n0 + cc] = (h16)tanh_fast(acc[i] + p1b[n0 + cc]);
  }
  __syncthreads();
  // decision = tanh(u@de+b), K=128
  for (int nt = wid; nt < 16; nt += 4) {
    int n0 = nt << 4;
    v8f acc = {};
    for (int kk = 0; kk < 128; kk += 32)
      acc = wmma16(ldfrag(uH + lr * RP128 + kk + lko), ldfrag(deT + (size_t)(n0 + lr) * 128 + kk + lko), acc);
#pragma unroll
    for (int i = 0; i < 8; ++i) decH[(rb + i) * RP256 + n0 + cc] = (h16)tanh_fast(acc[i] + deb[n0 + cc]);
  }
  // preds = v@p2+b, K=128, N=64 -> d_out [B,T,64]
  for (int nt = wid; nt < 4; nt += 4) {
    int n0 = nt << 4;
    v8f acc = {};
    for (int kk = 0; kk < 128; kk += 32)
      acc = wmma16(ldfrag(vH + lr * RP128 + kk + lko), ldfrag(p2T + (size_t)(n0 + lr) * 128 + kk + lko), acc);
#pragma unroll
    for (int i = 0; i < 8; ++i)
      preds[((size_t)(b0 + rb + i) * TSEQ + t) * 64 + n0 + cc] = acc[i] + p2b[n0 + cc];
  }
  __syncthreads();
  // actions = decision@oh+b, K=256, N=64
  for (int nt = wid; nt < 4; nt += 4) {
    int n0 = nt << 4;
    v8f acc = {};
    for (int kk = 0; kk < 256; kk += 32)
      acc = wmma16(ldfrag(decH + lr * RP256 + kk + lko), ldfrag(ohT + (size_t)(n0 + lr) * 256 + kk + lko), acc);
#pragma unroll
    for (int i = 0; i < 8; ++i)
      actions[((size_t)(b0 + rb + i) * TSEQ + t) * 64 + n0 + cc] = acc[i] + ohb[n0 + cc];
  }
}

extern "C" void kernel_launch(void* const* d_in, const int* in_sizes, int n_in,
                              void* d_out, int out_size, void* d_ws, size_t ws_size,
                              hipStream_t stream) {
  (void)in_sizes; (void)n_in; (void)out_size; (void)ws_size;
  const float* obs = (const float*)d_in[0];
#define P(i) ((const float*)d_in[i])

  char* ws = (char*)d_ws;
  size_t off = 0;
  auto carve = [&](size_t elems) -> h16* {
    h16* p = (h16*)(ws + off);
    off += ((elems * sizeof(h16) + 255) & ~(size_t)255);
    return p;
  };
  h16 *WeT = carve(256 * 64), *WcT = carve(128 * 256);
  h16 *WzT = carve(256 * 512), *WrT = carve(256 * 512), *WhT = carve(256 * 512);
  h16 *WpT = carve(256 * 256), *phiT = carve(768 * 768);
  h16 *ScT = carve(128 * 768), *SeT = carve(256 * 128);
  h16 *McT = carve(128 * 768), *MeT = carve(256 * 128);
  h16 *DcT = carve(128 * 768), *DeT = carve(256 * 128);
  h16 *dcT = carve(128 * 768), *deT = carve(256 * 128);
  h16 *ohT = carve(64 * 256), *p1T = carve(128 * 512), *p2T = carve(64 * 128);
  size_t actElems = (size_t)TSEQ * 64 * 256;
  h16 *X = carve(actElems), *SIN = carve(actElems), *Mbuf = carve(actElems);
  h16 *SSEQ = carve(actElems), *M2SEQ = carve(actElems), *DSEQ = carve(actElems);

  auto cv = [&](const float* s, h16* d, int K, int N) {
    int tot = K * N;
    k_wcvt<<<(tot + 255) / 256, 256, 0, stream>>>(s, d, K, N);
  };
  cv(P(1), WeT, 64, 256);   cv(P(3), WcT, 256, 128);
  cv(P(5), WzT, 512, 256);  cv(P(7), WrT, 512, 256);  cv(P(9), WhT, 512, 256);
  cv(P(11), WpT, 256, 256); cv(P(13), phiT, 768, 768);
  cv(P(15), ScT, 768, 128); cv(P(17), SeT, 128, 256);
  cv(P(19), McT, 768, 128); cv(P(21), MeT, 128, 256);
  cv(P(23), DcT, 768, 128); cv(P(25), DeT, 128, 256);
  cv(P(27), dcT, 768, 128); cv(P(29), deT, 128, 256);
  cv(P(31), ohT, 256, 64);  cv(P(33), p1T, 512, 128); cv(P(35), p2T, 128, 64);

  k_encode<<<8192, 128, 0, stream>>>(obs, WeT, P(2), WcT, P(4), X, SIN);
  k_gru<<<1, 256, GRU_SMEM, stream>>>(X, WzT, P(6), WrT, P(8), WhT, P(10), WpT, P(12), Mbuf);

  float* out = (float*)d_out;
  float* actions = out;
  float* preds = out + (size_t)BATCH * TSEQ * 64;
  float* outS = preds + (size_t)BATCH * TSEQ * 64;
  float* outM = outS + 64 * 256;
  float* outD = outM + 64 * 256;

  k_inter<<<1, 256, INT_SMEM, stream>>>(SIN, Mbuf, phiT, P(14), ScT, P(16), SeT, P(18),
                                        McT, P(20), MeT, P(22), DcT, P(24), DeT, P(26),
                                        SSEQ, M2SEQ, DSEQ, outS, outM, outD);
  k_post<<<8192, 128, 0, stream>>>(SSEQ, M2SEQ, DSEQ, dcT, P(28), deT, P(30),
                                   ohT, P(32), p1T, P(34), p2T, P(36), actions, preds);
#undef P
}
